// BahdanauAttention_50036368998583
// MI455X (gfx1250) — matrix operand order repaired and checked by reading, not compiled, not run on hardware
//
#include <hip/hip_runtime.h>
#include <hip/hip_bf16.h>

// ---------------------------------------------------------------------------
// Bahdanau attention for MI455X (gfx1250, wave32, WMMA).
// B=64, S=2048, H=512.
// Dominant GEMM (keys @ U_w^T, 68.7 GFLOP) runs on v_wmma_f32_16x16x32_bf16,
// fused with tanh + v-dot so the [B,S,H] energy tensor never touches HBM.
// ---------------------------------------------------------------------------

#define BATCH 64
#define SEQ   2048
#define HID   512
#define NROWS (BATCH * SEQ)        // 131072 flattened (b,s) rows
#define SROW  520                  // padded LDS row stride (bf16 elements)

typedef __attribute__((ext_vector_type(16))) __bf16 v16bf;
typedef __attribute__((ext_vector_type(8)))  float  v8f;

union TileReg {
    uint4 q[2];   // two 16-byte chunks (VGPRs 0-3 / 4-7)
    v16bf v;      // 16 bf16 per lane
};

__device__ __forceinline__ unsigned short f2bf(float x) {
    unsigned u = __float_as_uint(x);
    unsigned r = (u + 0x7FFFu + ((u >> 16) & 1u)) >> 16;   // RNE
    return (unsigned short)r;
}

// ---------------------------------------------------------------------------
// Kernel 1: Wq[b,g] = sum_h query[b,h] * W_w[g,h] + W_b[g]
// ---------------------------------------------------------------------------
__global__ __launch_bounds__(256) void k_wq(const float* __restrict__ query,
                                            const float* __restrict__ Ww,
                                            const float* __restrict__ Wb,
                                            float* __restrict__ wq) {
    __shared__ float qs[HID];
    const int b = blockIdx.x;
    const int tid = threadIdx.x;
    qs[tid]       = query[b * HID + tid];
    qs[tid + 256] = query[b * HID + tid + 256];
    __syncthreads();
    const float4* qv = (const float4*)qs;
#pragma unroll
    for (int gi = 0; gi < 2; ++gi) {
        const int g = tid + gi * 256;
        const float4* wr = (const float4*)(Ww + (size_t)g * HID);
        float acc = Wb[g];
        for (int k = 0; k < HID / 4; ++k) {
            float4 a = qv[k], w = wr[k];
            acc += a.x * w.x + a.y * w.y + a.z * w.z + a.w * w.w;
        }
        wq[b * HID + g] = acc;
    }
}

// ---------------------------------------------------------------------------
// Kernel 2: pre-swizzle U_w (fp32, [g][h]) into bf16 WMMA B-matrix layout.
// B-tile (32x16 KxN, 8 VGPRs): lane l: N = l&15, K = (v<4?0:16) + (l<16?0:8) + 2(v&3),+1.
// Stored as uint4 chunks: chunk index = ((j*16+t)*2+c)*32 + l  (c = VGPR 0-3/4-7)
// so the consumer does one coalesced b128 load per chunk.
// ---------------------------------------------------------------------------
__global__ __launch_bounds__(256) void k_bpack(const float* __restrict__ Uw,
                                               unsigned int* __restrict__ bpack) {
    const int idx = blockIdx.x * 256 + threadIdx.x;      // 131072 dwords
    const int v = idx & 3;
    const int l = (idx >> 2) & 31;
    const int c = (idx >> 7) & 1;
    const int rest = idx >> 8;
    const int t = rest & 15;        // k-step (32 elements each)
    const int j = rest >> 4;        // n-tile (16 cols each)
    const int g  = j * 16 + (l & 15);
    const int h0 = t * 32 + ((l >> 4) * 8) + c * 16 + 2 * v;
    const unsigned lo = f2bf(Uw[(size_t)g * HID + h0]);
    const unsigned hi = f2bf(Uw[(size_t)g * HID + h0 + 1]);
    bpack[idx] = lo | (hi << 16);
}

// ---------------------------------------------------------------------------
// Kernel 3: fused energy GEMM + tanh + v-dot -> attention logits [NROWS].
// Block = 256 threads (8 waves), covers 32 rows x 512 cols.
// Wave w: m-tiles {0,1}, n-tiles {4w .. 4w+3}.  K = 512 in 16 steps of 32.
// ---------------------------------------------------------------------------
__global__ __launch_bounds__(256) void k_energy(const float* __restrict__ keys,
                                                const float* __restrict__ wq,
                                                const float* __restrict__ Ub,
                                                const float* __restrict__ vw,
                                                const float* __restrict__ vb,
                                                const uint4* __restrict__ bpack,
                                                float* __restrict__ logits) {
    __shared__ unsigned short sk[32 * SROW];   // bf16 keys tile, padded rows
    __shared__ float red[8 * 32];              // per-wave row partials

    const int tid  = threadIdx.x;
    const int wv   = tid >> 5;                 // wave 0..7
    const int l    = tid & 31;                 // lane
    const int half = l >> 4;
    const size_t r0 = (size_t)blockIdx.x * 32; // first flattened row
    const int bidx = (int)(r0 >> 11);          // batch index (rows never cross b)

    // ---- stage 32x512 fp32 keys rows -> bf16 LDS (coalesced float4 loads) ----
    const float* kbase = keys + r0 * HID;
#pragma unroll
    for (int i = 0; i < 16; ++i) {
        const int f = (i * 256 + tid) * 4;     // 0..16383
        const int row = f >> 9, col = f & 511;
        float4 x = *(const float4*)(kbase + f);
        unsigned p0 = (unsigned)f2bf(x.x) | ((unsigned)f2bf(x.y) << 16);
        unsigned p1 = (unsigned)f2bf(x.z) | ((unsigned)f2bf(x.w) << 16);
        *(uint2*)(&sk[row * SROW + col]) = make_uint2(p0, p1);
    }
    __syncthreads();

    v8f acc[2][4];
#pragma unroll
    for (int mt = 0; mt < 2; ++mt)
#pragma unroll
        for (int jt = 0; jt < 4; ++jt) acc[mt][jt] = (v8f)0.0f;

    // ---- K loop: 16 steps of K=32 ----
    for (int t = 0; t < 16; ++t) {
        TileReg a[2];
        const int koff = t * 32 + half * 8;    // A layout: lanes>=16 hold K+8
#pragma unroll
        for (int mt = 0; mt < 2; ++mt) {
            const int ro = (mt * 16 + (l & 15)) * SROW + koff;
            a[mt].q[0] = *(const uint4*)(&sk[ro]);        // K rel 0-7  (+8)
            a[mt].q[1] = *(const uint4*)(&sk[ro + 16]);   // K rel 16-23(+8)
        }
#pragma unroll
        for (int jt = 0; jt < 4; ++jt) {
            const int jg = wv * 4 + jt;
            TileReg bm;
            const int cbase = ((jg * 16 + t) * 2) * 32 + l;
            bm.q[0] = bpack[cbase];
            bm.q[1] = bpack[cbase + 32];
            acc[0][jt] = __builtin_amdgcn_wmma_f32_16x16x32_bf16(
                false, a[0].v, false, bm.v, (short)0, acc[0][jt], false, false);
            acc[1][jt] = __builtin_amdgcn_wmma_f32_16x16x32_bf16(
                false, a[1].v, false, bm.v, (short)0, acc[1][jt], false, false);
        }
    }

    // ---- epilogue: tanh(acc + Wq + Ub) . v_w, reduce over columns ----
    float wb[4], vvw[4];
#pragma unroll
    for (int jt = 0; jt < 4; ++jt) {
        const int col = (wv * 4 + jt) * 16 + (l & 15);
        wb[jt]  = wq[bidx * HID + col] + Ub[col];
        vvw[jt] = vw[col];
    }
    float s[2][8];
#pragma unroll
    for (int mt = 0; mt < 2; ++mt)
#pragma unroll
        for (int r = 0; r < 8; ++r) {
            float acc_s = 0.0f;
#pragma unroll
            for (int jt = 0; jt < 4; ++jt)
                acc_s += vvw[jt] * tanhf(acc[mt][jt][r] + wb[jt]);
            s[mt][r] = acc_s;
        }
    // butterfly across the 16 lanes sharing the same row half
#pragma unroll
    for (int mt = 0; mt < 2; ++mt)
#pragma unroll
        for (int r = 0; r < 8; ++r) {
#pragma unroll
            for (int d = 1; d < 16; d <<= 1)
                s[mt][r] += __shfl_xor(s[mt][r], d, 16);
        }
    // C/D layout: VGPR r, lanes<16 -> M=r, lanes>=16 -> M=r+8
    const int k = l & 15;
    red[wv * 32 + (k >> 3) * 16 + half * 8 + (k & 7)] = s[k >> 3][k & 7];
    __syncthreads();
    if (tid < 32) {
        float tot = vb[0];
#pragma unroll
        for (int w2 = 0; w2 < 8; ++w2) tot += red[w2 * 32 + tid];
        logits[r0 + tid] = tot;
    }
}

// ---------------------------------------------------------------------------
// Kernel 4: softmax over S per batch row -> attention weights (output).
// ---------------------------------------------------------------------------
__global__ __launch_bounds__(256) void k_softmax(const float* __restrict__ logits,
                                                 float* __restrict__ outw) {
    __shared__ float red[256];
    const int b = blockIdx.x, tid = threadIdx.x;
    const float* row = logits + (size_t)b * SEQ;
    float x[8];
    float m = -3.402823466e38f;
#pragma unroll
    for (int i = 0; i < 8; ++i) { x[i] = row[tid + i * 256]; m = fmaxf(m, x[i]); }
    red[tid] = m; __syncthreads();
    for (int st = 128; st > 0; st >>= 1) {
        if (tid < st) red[tid] = fmaxf(red[tid], red[tid + st]);
        __syncthreads();
    }
    m = red[0]; __syncthreads();
    float sum = 0.0f;
#pragma unroll
    for (int i = 0; i < 8; ++i) { x[i] = __expf(x[i] - m); sum += x[i]; }
    red[tid] = sum; __syncthreads();
    for (int st = 128; st > 0; st >>= 1) {
        if (tid < st) red[tid] += red[tid + st];
        __syncthreads();
    }
    const float inv = 1.0f / red[0];
#pragma unroll
    for (int i = 0; i < 8; ++i) outw[(size_t)b * SEQ + tid + i * 256] = x[i] * inv;
}

// ---------------------------------------------------------------------------
// Kernel 5/6: context[b,h] = sum_s w[b,s]*keys[b,s,h] (deterministic 2-stage).
// ---------------------------------------------------------------------------
__global__ __launch_bounds__(256) void k_ctx_part(const float* __restrict__ keys,
                                                  const float* __restrict__ attw,
                                                  float* __restrict__ part) {
    __shared__ float wsh[256];
    const int b = blockIdx.x, sc = blockIdx.y, tid = threadIdx.x;
    wsh[tid] = attw[(size_t)b * SEQ + sc * 256 + tid];
    __syncthreads();
    const float* kb = keys + ((size_t)b * SEQ + sc * 256) * HID;
    float a0 = 0.0f, a1 = 0.0f;
    for (int s = 0; s < 256; ++s) {
        const float wvv = wsh[s];
        a0 += wvv * kb[(size_t)s * HID + tid];
        a1 += wvv * kb[(size_t)s * HID + tid + 256];
    }
    part[(size_t)(b * 8 + sc) * HID + tid]       = a0;
    part[(size_t)(b * 8 + sc) * HID + tid + 256] = a1;
}

__global__ __launch_bounds__(256) void k_ctx_reduce(const float* __restrict__ part,
                                                    float* __restrict__ ctx) {
    const int idx = blockIdx.x * 256 + threadIdx.x;   // 32768 = b*512+h
    const int b = idx >> 9, h = idx & 511;
    float tot = 0.0f;
#pragma unroll
    for (int sc = 0; sc < 8; ++sc) tot += part[(size_t)(b * 8 + sc) * HID + h];
    ctx[idx] = tot;
}

// ---------------------------------------------------------------------------
// Workspace layout (bytes):
//   [0,       131072) Wq        fp32 [64][512]
//   [131072,  655360) Bpack     bf16-swizzled U_w (uint32 x 131072)
//   [655360, 1179648) logits    fp32 [131072]
//   [1179648,2228224) ctx part  fp32 [64][8][512]
// ---------------------------------------------------------------------------
extern "C" void kernel_launch(void* const* d_in, const int* in_sizes, int n_in,
                              void* d_out, int out_size, void* d_ws, size_t ws_size,
                              hipStream_t stream) {
    (void)in_sizes; (void)n_in; (void)out_size; (void)ws_size;
    const float* query = (const float*)d_in[0];
    const float* keys  = (const float*)d_in[1];
    const float* Ww    = (const float*)d_in[2];
    const float* Wb    = (const float*)d_in[3];
    const float* Uw    = (const float*)d_in[4];
    const float* Ub    = (const float*)d_in[5];
    const float* vw    = (const float*)d_in[6];
    const float* vb    = (const float*)d_in[7];

    char* ws = (char*)d_ws;
    float*        wq     = (float*)(ws + 0);
    unsigned int* bpack  = (unsigned int*)(ws + 131072);
    float*        logits = (float*)(ws + 655360);
    float*        part   = (float*)(ws + 1179648);

    float* ctx  = (float*)d_out;                    // [64][512]
    float* attw = (float*)d_out + BATCH * HID;      // [64][2048]

    k_wq<<<BATCH, 256, 0, stream>>>(query, Ww, Wb, wq);
    k_bpack<<<512, 256, 0, stream>>>(Uw, bpack);
    k_energy<<<NROWS / 32, 256, 0, stream>>>(keys, wq, Ub, vw, vb,
                                             (const uint4*)bpack, logits);
    k_softmax<<<BATCH, 256, 0, stream>>>(logits, attw);
    k_ctx_part<<<dim3(BATCH, 8), 256, 0, stream>>>(keys, attw, part);
    k_ctx_reduce<<<BATCH * HID / 256, 256, 0, stream>>>(part, ctx);
}
